// KoLeoLoss_51092930953341
// MI455X (gfx1250) — matrix operand order repaired and checked by reading, not compiled, not run on hardware
//
#include <hip/hip_runtime.h>
#include <hip/hip_bf16.h>

// ---------------------------------------------------------------------------
// KoLeo loss for MI455X (gfx1250, wave32, WMMA)
//   B = 8192 rows, D = 512 features.
//   GEMM (8192x8192x512, 68.7 GFLOP) via V_WMMA_F32_16X16X32_F16 (compute-
//   bound; all data fits in 192MB L2, HBM traffic ~25MB ≈ 1us @ 23.3TB/s).
//   B tiles staged with GLOBAL_LOAD_ASYNC_TO_LDS_B128 (double-buffered,
//   ASYNCcnt-tracked DMA); A fragments are register-resident.
//   Final distances recomputed in f32 from raw input + stored 1/norm.
// ---------------------------------------------------------------------------

#define B_N 8192
#define D_K 512
#define NCHUNK 16                    // column chunks for the argmax GEMM
#define CHUNK_COLS (B_N / NCHUNK)    // 512 columns per chunk
#define NTILES (CHUNK_COLS / 16)     // 32 16-column tiles per chunk
#define ROWS_PER_BLOCK 128           // 8 waves * 16 rows
#define KOLEO_EPS 1e-8f

#if defined(__has_builtin)
#  if __has_builtin(__builtin_amdgcn_global_load_async_to_lds_b128) && \
      __has_builtin(__builtin_amdgcn_s_wait_asynccnt)
#    define KOLEO_ASYNC 1
#  endif
#endif
#ifndef KOLEO_ASYNC
#  define KOLEO_ASYNC 0
#endif

typedef __attribute__((ext_vector_type(16))) _Float16 v16h;
typedef __attribute__((ext_vector_type(8)))  _Float16 v8h;
typedef __attribute__((ext_vector_type(8)))  float    v8f;

// Builtin signature (from hipcc diagnostic): V4i*1, V4i*3, Ii, Ii
typedef int v4i_b128 __attribute__((vector_size(16)));
typedef __attribute__((address_space(1))) v4i_b128* koleo_gptr128;
typedef __attribute__((address_space(3))) v4i_b128* koleo_lptr128;

union AFrag { v16h v; v8h h[2]; };

// Stage one 16-column x 512-K f16 tile (16 KB) into LDS.
// Async path: 4 GLOBAL_LOAD_ASYNC_TO_LDS_B128 per thread (ASYNCcnt-tracked
// DMA, no VGPR round trip). Fallback: plain b128 copy.
__device__ __forceinline__ void koleo_stage_tile(const _Float16* __restrict__ xh,
                                                 int colBase,
                                                 _Float16* dst, int tid) {
    #pragma unroll
    for (int q = tid; q < 1024; q += 256) {
        const int c   = q >> 6;        // column within tile
        const int k16 = q & 63;        // 16-byte chunk along K
        const _Float16* src = xh + (size_t)(colBase + c) * D_K + k16 * 8;
#if KOLEO_ASYNC
        __builtin_amdgcn_global_load_async_to_lds_b128(
            (koleo_gptr128)src, (koleo_lptr128)(dst + q * 8), 0, 0);
#else
        *(uint4*)(dst + q * 8) = *(const uint4*)src;
#endif
    }
}

// ---------------------------------------------------------------------------
// Kernel 1: row L2-normalize; emit f16 matrix for WMMA and 1/norm per row.
// ---------------------------------------------------------------------------
__global__ void __launch_bounds__(256)
koleo_normalize_kernel(const float* __restrict__ s,
                       _Float16* __restrict__ xh,
                       float* __restrict__ invnorm) {
    const int row = blockIdx.x;
    const int tid = threadIdx.x;
    const float* sr = s + (size_t)row * D_K;

    float a = sr[tid];
    float b = sr[tid + 256];
    float ss = a * a + b * b;

    #pragma unroll
    for (int off = 16; off >= 1; off >>= 1) ss += __shfl_xor(ss, off, 32);

    __shared__ float red[8];
    __shared__ float invs;
    if ((tid & 31) == 0) red[tid >> 5] = ss;
    __syncthreads();
    if (tid == 0) {
        float t = 0.f;
        #pragma unroll
        for (int w = 0; w < 8; ++w) t += red[w];
        float inv = 1.0f / fmaxf(sqrtf(t), KOLEO_EPS);
        invnorm[row] = inv;
        invs = inv;
    }
    __syncthreads();
    const float inv = invs;
    xh[(size_t)row * D_K + tid]       = (_Float16)(a * inv);
    xh[(size_t)row * D_K + tid + 256] = (_Float16)(b * inv);
}

// ---------------------------------------------------------------------------
// Kernel 2: dots = x x^T via WMMA f16 -> f32, fused per-row running argmax.
// Grid: (NCHUNK, B_N/ROWS_PER_BLOCK). Block: 256 threads = 8 waves.
// Double-buffered LDS B tiles; two independent WMMA accumulation chains.
// ---------------------------------------------------------------------------
__global__ void __launch_bounds__(256)
koleo_gemm_argmax_kernel(const _Float16* __restrict__ xh,
                         float* __restrict__ pMax,
                         int* __restrict__ pIdx) {
    __shared__ _Float16 ldsB[2][16 * D_K];   // 2 x 16 KB double buffer

    const int tid   = threadIdx.x;
    const int wave  = tid >> 5;
    const int lid   = tid & 31;              // lane in wave32
    const int chunk = blockIdx.x;
    const int rowBase = blockIdx.y * ROWS_PER_BLOCK + wave * 16;
    const int colChunkBase = chunk * CHUNK_COLS;

    // Fragment layouts used here (contiguous k, same order in A and B; not the ISA table order):
    //  A (16x32 f16): lane m<16 -> row m, K 0..15; lane m+16 -> K 16..31
    //  B (32x16 f16): lanes 0-15 -> K 0..15, lanes 16-31 -> K 16..31; lane%16 = N
    const int aRow = rowBase + (lid & 15);
    const int aOff = (lid < 16) ? 0 : 16;
    const int bOff = (lid < 16) ? 0 : 16;
    const int bCol = lid & 15;
    const int rowHalf = (lid >> 4) * 8;      // C/D layout: lanes 16-31 hold M=8..15

    float bestVal[8];
    int   bestIdx[8];
    #pragma unroll
    for (int v = 0; v < 8; ++v) { bestVal[v] = -3.0e38f; bestIdx[v] = 0; }

    // Preload tile 0.
    koleo_stage_tile(xh, colChunkBase, ldsB[0], tid);
#if KOLEO_ASYNC
    __builtin_amdgcn_s_wait_asynccnt(0);
#endif
    __syncthreads();

    int buf = 0;
    for (int nt = 0; nt < NTILES; ++nt) {
        const int colBase = colChunkBase + nt * 16;

        // Kick off DMA of the next tile into the other buffer.
        if (nt + 1 < NTILES)
            koleo_stage_tile(xh, colBase + 16, ldsB[buf ^ 1], tid);

        const _Float16* bsrc = ldsB[buf];

        // Two independent accumulation chains over K = 512.
        v8f acc0 = {};
        v8f acc1 = {};
        #pragma unroll
        for (int k0 = 0; k0 < D_K; k0 += 64) {
            AFrag a0, a1;
            const _Float16* ap = xh + (size_t)aRow * D_K + k0 + aOff;
            a0.h[0] = *(const v8h*)(ap);
            a0.h[1] = *(const v8h*)(ap + 8);
            a1.h[0] = *(const v8h*)(ap + 32);
            a1.h[1] = *(const v8h*)(ap + 40);
            v16h b0 = *(const v16h*)&bsrc[bCol * D_K + k0 + bOff];
            v16h b1 = *(const v16h*)&bsrc[bCol * D_K + k0 + 32 + bOff];
            acc0 = __builtin_amdgcn_wmma_f32_16x16x32_f16(
                       false, a0.v, false, b0, (short)0, acc0, false, false);
            acc1 = __builtin_amdgcn_wmma_f32_16x16x32_f16(
                       false, a1.v, false, b1, (short)0, acc1, false, false);
        }
        v8f acc = acc0 + acc1;

        const int col = colBase + bCol;
        #pragma unroll
        for (int v = 0; v < 8; ++v) {
            float val = acc[v];
            const int grow = rowBase + rowHalf + v;
            if (grow == col) val = -1.0f;          // mask diagonal (self-match)

            // max+argmax across the 16 lanes of this half-group
            float mval = val;
            int   midx = col;
            #pragma unroll
            for (int off = 8; off >= 1; off >>= 1) {
                float ov = __shfl_xor(mval, off, 16);
                int   oi = __shfl_xor(midx, off, 16);
                if (ov > mval || (ov == mval && oi < midx)) { mval = ov; midx = oi; }
            }
            // columns ascend across nt; strict > keeps first-max semantics
            if (mval > bestVal[v]) { bestVal[v] = mval; bestIdx[v] = midx; }
        }

        // One barrier: (a) all waves finished reading ldsB[buf] before it is
        // overwritten next iteration, (b) next tile's async DMA (waited just
        // below) is visible to every wave.
#if KOLEO_ASYNC
        if (nt + 1 < NTILES) __builtin_amdgcn_s_wait_asynccnt(0);
#endif
        __syncthreads();
        buf ^= 1;
    }

    if ((lid & 15) == 0) {
        #pragma unroll
        for (int v = 0; v < 8; ++v) {
            const int grow = rowBase + rowHalf + v;
            pMax[(size_t)grow * NCHUNK + chunk] = bestVal[v];
            pIdx[(size_t)grow * NCHUNK + chunk] = bestIdx[v];
        }
    }
}

// ---------------------------------------------------------------------------
// Kernel 3: combine the NCHUNK chunk-partials into the final NN index.
// ---------------------------------------------------------------------------
__global__ void __launch_bounds__(256)
koleo_argmax_reduce_kernel(const float* __restrict__ pMax,
                           const int* __restrict__ pIdx,
                           int* __restrict__ I) {
    const int row = blockIdx.x * blockDim.x + threadIdx.x;
    if (row >= B_N) return;
    float bv = -3.0e38f;
    int   bi = 0;
    #pragma unroll
    for (int c = 0; c < NCHUNK; ++c) {
        const float v = pMax[(size_t)row * NCHUNK + c];
        const int   i = pIdx[(size_t)row * NCHUNK + c];
        if (v > bv || (v == bv && i < bi)) { bv = v; bi = i; }
    }
    I[row] = bi;
}

// ---------------------------------------------------------------------------
// Kernel 4: f32 pairwise distance to NN; per-row log term.
// ---------------------------------------------------------------------------
__global__ void __launch_bounds__(256)
koleo_distance_kernel(const float* __restrict__ s,
                      const float* __restrict__ invnorm,
                      const int* __restrict__ I,
                      float* __restrict__ rowLoss) {
    const int row = blockIdx.x;
    const int tid = threadIdx.x;
    const int j = I[row];
    const float invi = invnorm[row];
    const float invj = invnorm[j];
    const float* si = s + (size_t)row * D_K;
    const float* sj = s + (size_t)j * D_K;

    float a0 = si[tid] * invi - sj[tid] * invj + KOLEO_EPS;
    float a1 = si[tid + 256] * invi - sj[tid + 256] * invj + KOLEO_EPS;
    float ss = a0 * a0 + a1 * a1;

    #pragma unroll
    for (int off = 16; off >= 1; off >>= 1) ss += __shfl_xor(ss, off, 32);

    __shared__ float red[8];
    if ((tid & 31) == 0) red[tid >> 5] = ss;
    __syncthreads();
    if (tid == 0) {
        float t = 0.f;
        #pragma unroll
        for (int w = 0; w < 8; ++w) t += red[w];
        rowLoss[row] = logf(sqrtf(t) + KOLEO_EPS);
    }
}

// ---------------------------------------------------------------------------
// Kernel 5: loss = -mean(rowLoss)
// ---------------------------------------------------------------------------
__global__ void __launch_bounds__(256)
koleo_final_reduce_kernel(const float* __restrict__ rowLoss,
                          float* __restrict__ out) {
    const int tid = threadIdx.x;
    float acc = 0.f;
    for (int i = tid; i < B_N; i += 256) acc += rowLoss[i];
    #pragma unroll
    for (int off = 16; off >= 1; off >>= 1) acc += __shfl_xor(acc, off, 32);
    __shared__ float red[8];
    if ((tid & 31) == 0) red[tid >> 5] = acc;
    __syncthreads();
    if (tid == 0) {
        float t = 0.f;
        #pragma unroll
        for (int w = 0; w < 8; ++w) t += red[w];
        out[0] = -t / (float)B_N;
    }
}

// ---------------------------------------------------------------------------
extern "C" void kernel_launch(void* const* d_in, const int* in_sizes, int n_in,
                              void* d_out, int out_size, void* d_ws, size_t ws_size,
                              hipStream_t stream) {
    (void)in_sizes; (void)n_in; (void)out_size; (void)ws_size;

    const float* s = (const float*)d_in[0];
    float* out = (float*)d_out;

    char* ws = (char*)d_ws;
    size_t off = 0;
    auto take = [&](size_t bytes) {
        char* p = ws + off;
        off += (bytes + 255) & ~(size_t)255;
        return (void*)p;
    };
    _Float16* xh      = (_Float16*)take((size_t)B_N * D_K * sizeof(_Float16)); // 8 MB
    float*    invnorm = (float*)take((size_t)B_N * sizeof(float));
    float*    pMax    = (float*)take((size_t)B_N * NCHUNK * sizeof(float));
    int*      pIdx    = (int*)take((size_t)B_N * NCHUNK * sizeof(int));
    int*      I       = (int*)take((size_t)B_N * sizeof(int));
    float*    rowLoss = (float*)take((size_t)B_N * sizeof(float));

    koleo_normalize_kernel<<<B_N, 256, 0, stream>>>(s, xh, invnorm);

    dim3 g2(NCHUNK, B_N / ROWS_PER_BLOCK);
    koleo_gemm_argmax_kernel<<<g2, 256, 0, stream>>>(xh, pMax, pIdx);

    koleo_argmax_reduce_kernel<<<(B_N + 255) / 256, 256, 0, stream>>>(pMax, pIdx, I);
    koleo_distance_kernel<<<B_N, 256, 0, stream>>>(s, invnorm, I, rowLoss);
    koleo_final_reduce_kernel<<<1, 256, 0, stream>>>(rowLoss, out);
}
